// CausalSelfAttention_52845277610031
// MI455X (gfx1250) — matrix operand order repaired and checked by reading, not compiled, not run on hardware
//
#include <hip/hip_runtime.h>
#include <hip/hip_bf16.h>
#include <stdint.h>

// ---------------------------------------------------------------------------
// Causal self-attention for MI455X (gfx1250), wave32 WMMA f16->f32 pipeline.
//   B=4, T=2048, C=1024, H=16, D=64
// Stages: cast/transpose -> QKV GEMM -> flash attention -> proj GEMM
// GEMM wave tile widened to 32M x 64N: 8 wmma per 32-K step, B-frag reuse x2.
// ---------------------------------------------------------------------------

typedef _Float16 h16;
typedef __attribute__((ext_vector_type(16))) _Float16 v16h;
typedef __attribute__((ext_vector_type(8)))  float    v8f;

#define BB   4
#define TT   2048
#define CC   1024
#define HH   16
#define DD   64
#define BT   (BB * TT)      // 8192
#define C3   (3 * CC)       // 3072

__device__ __forceinline__ v8f wmma16(v16h a, v16h b, v8f c) {
  // D = A(16x32 f16) x B(32x16 f16) + C(16x16 f32)
  return __builtin_amdgcn_wmma_f32_16x16x32_f16(
      /*neg_a=*/false, a, /*neg_b=*/false, b,
      /*c_mod=*/(short)0, c, /*reuse_a=*/false, /*reuse_b=*/false);
}

union FragA { v16h v; uint32_t u[8]; };
union FragB { v16h v; uint4 q[2]; };

// A-matrix 16x32 fragment from row-major [.., lda] f16 at (row0, k0).
// Lane l holds M = row0 + (l&15); half=l>>4 selects K 0..15 vs 16..31.
// Compiler merges these into two 16B global_load_b128 per fragment.
__device__ __forceinline__ v16h load_a(const h16* __restrict__ A, int lda,
                                       int row0, int k0, int lane) {
  FragA f;
  const int hf = lane >> 4;
  const h16* base = A + (size_t)(row0 + (lane & 15)) * lda + k0;
#pragma unroll
  for (int g = 0; g < 8; ++g) {
    int kk = 2 * g + 16 * hf;                       // K pair start
    f.u[g] = *reinterpret_cast<const uint32_t*>(base + kk);
  }
  return f.v;
}

// B-matrix 32x16 fragment from Bt = B^T stored [N, K] row-major (K contiguous).
// Lane l holds N = n0 + (l&15); half selects K 0..15 vs 16..31.
// Requires ldb % 8 == 0 and k0 % 32 == 0 -> 16B-aligned b128 loads.
__device__ __forceinline__ v16h load_b(const h16* __restrict__ Bt, int ldb,
                                       int n0, int k0, int lane) {
  FragB f;
  const h16* base = Bt + (size_t)(n0 + (lane & 15)) * ldb + k0 + 16 * (lane >> 4);
  f.q[0] = *reinterpret_cast<const uint4*>(base);
  f.q[1] = *reinterpret_cast<const uint4*>(base + 8);
  return f.v;
}

// ---------------------------------------------------------------------------
// Stage 1: casts
// ---------------------------------------------------------------------------
__global__ void k_cast_f32_f16(const float* __restrict__ in,
                               h16* __restrict__ out, int n) {
  int i = blockIdx.x * 256 + threadIdx.x;
  if (i < n) out[i] = (h16)in[i];
}

// out[c * rows + r] = (h16) in[r * cols + c]   (store W^T so K is contiguous)
__global__ void k_transpose_cast(const float* __restrict__ in,
                                 h16* __restrict__ out, int rows, int cols) {
  int i = blockIdx.x * 256 + threadIdx.x;
  if (i >= rows * cols) return;
  int r = i / cols, c = i % cols;               // contiguous global reads
  out[(size_t)c * rows + r] = (h16)in[i];
}

// ---------------------------------------------------------------------------
// Stage 2: QKV GEMM.  [8192,1024] x [1024,3072] -> scatter into Q,K [BH,T,64]
// and Vt [BH,64,T].  Block = 8 waves (2M x 4N), wave tile = 32M x 64N.
// ---------------------------------------------------------------------------
__global__ __launch_bounds__(256) void k_gemm_qkv(
    const h16* __restrict__ Xh, const h16* __restrict__ Wt,
    h16* __restrict__ Qo, h16* __restrict__ Ko, h16* __restrict__ Vto) {
  const int lane = threadIdx.x & 31, wave = threadIdx.x >> 5;
  const int hf = lane >> 4, ln = lane & 15;
  const int m0 = blockIdx.y * 64  + (wave >> 2) * 32;
  const int n0 = blockIdx.x * 256 + (wave & 3) * 64;

  v8f acc[2][4] = {{v8f{}, v8f{}, v8f{}, v8f{}}, {v8f{}, v8f{}, v8f{}, v8f{}}};
  for (int k0 = 0; k0 < CC; k0 += 32) {
    if (k0 + 256 < CC) {   // speculative prefetch ~8 K-tiles ahead
      __builtin_prefetch(Xh + (size_t)(m0 + ln) * CC + k0 + 256, 0, 3);
      __builtin_prefetch(Xh + (size_t)(m0 + 16 + ln) * CC + k0 + 256, 0, 3);
      __builtin_prefetch(Wt + (size_t)(n0 + hf * 32 + ln) * CC + k0 + 256, 0, 3);
    }
    v16h a0 = load_a(Xh, CC, m0,      k0, lane);
    v16h a1 = load_a(Xh, CC, m0 + 16, k0, lane);
#pragma unroll
    for (int j = 0; j < 4; ++j) {
      v16h b = load_b(Wt, CC, n0 + j * 16, k0, lane);
      acc[0][j] = wmma16(a0, b, acc[0][j]);
      acc[1][j] = wmma16(a1, b, acc[1][j]);
    }
  }
#pragma unroll
  for (int mi = 0; mi < 2; ++mi) {
#pragma unroll
    for (int j = 0; j < 4; ++j) {
      int col = n0 + j * 16 + ln;
      int sect = col >> 10;                // 0=q 1=k 2=v
      int cc = col & (CC - 1);
      int h = cc >> 6, d = cc & 63;
#pragma unroll
      for (int r = 0; r < 8; ++r) {
        int row = m0 + mi * 16 + r + 8 * hf;  // row in [0, B*T)
        int b = row >> 11, t = row & (TT - 1);
        int bh = (b << 4) + h;
        h16 v = (h16)acc[mi][j][r];
        if (sect == 0)      Qo[((size_t)bh * TT + t) * DD + d] = v;
        else if (sect == 1) Ko[((size_t)bh * TT + t) * DD + d] = v;
        else                Vto[((size_t)bh * DD + d) * TT + t] = v;
      }
    }
  }
}

// ---------------------------------------------------------------------------
// Stage 3: flash attention.  One wave = one 16-row q tile of one (b,h).
// Keys processed 32 at a time: 4 WMMAs for S, online softmax (shfl-16 row
// reductions), P staged f32->f16 through LDS to re-layout C->A, 4 WMMAs PV.
// ---------------------------------------------------------------------------
__global__ __launch_bounds__(256) void k_attn(
    const h16* __restrict__ Q, const h16* __restrict__ Kc,
    const h16* __restrict__ Vt, h16* __restrict__ Y) {
  __shared__ _Float16 smem[8][16 * 32];    // per-wave 16x32 P tile (1 KB each)

  const int lane = threadIdx.x & 31, wave = threadIdx.x >> 5;
  const int hf = lane >> 4, ln = lane & 15;
  const int tile = blockIdx.x * 8 + wave;  // 8192 tiles total
  const int qt = tile & 127;               // T/16 = 128 q tiles
  const int bh = tile >> 7;
  const int qb = qt << 4;

  const h16* Qb = Q  + (size_t)bh * TT * DD;
  const h16* Kb = Kc + (size_t)bh * TT * DD;
  const h16* Vb = Vt + (size_t)bh * DD * TT;

  // Q fragments for this tile, reused for every key chunk.
  const v16h aq0 = load_a(Qb, DD, qb, 0, lane);
  const v16h aq1 = load_a(Qb, DD, qb, 32, lane);

  v8f o[4] = {v8f{}, v8f{}, v8f{}, v8f{}};
  float mrow[8], lrow[8];
#pragma unroll
  for (int r = 0; r < 8; ++r) { mrow[r] = -1e30f; lrow[r] = 0.0f; }
  const float scale = 0.125f;              // 1/sqrt(64)

  for (int kb = 0; kb < qb + 16; kb += 32) {
    if (kb + 32 < qb + 16) {               // prefetch next key chunk
      __builtin_prefetch(Kb + (size_t)(kb + 32 + lane) * DD, 0, 3);
      __builtin_prefetch(Vb + (size_t)(ln + 16 * hf) * TT + kb + 32, 0, 3);
    }
    // S = (Q K^T) tile pair, 16x32 keys
    v8f s0 = {}, s1 = {};
    s0 = wmma16(aq0, load_b(Kb, DD, kb,      0,  lane), s0);
    s0 = wmma16(aq1, load_b(Kb, DD, kb,      32, lane), s0);
    s1 = wmma16(aq0, load_b(Kb, DD, kb + 16, 0,  lane), s1);
    s1 = wmma16(aq1, load_b(Kb, DD, kb + 16, 32, lane), s1);

    // scale + causal mask (sentinel avoids inf-inf NaN)
#pragma unroll
    for (int r = 0; r < 8; ++r) {
      int qg = qb + r + 8 * hf;
      float a0 = s0[r] * scale; if (kb + ln      > qg) a0 = -1e30f;
      float a1 = s1[r] * scale; if (kb + 16 + ln > qg) a1 = -1e30f;
      s0[r] = a0; s1[r] = a1;
    }

    float corr[8], ps[8];
#pragma unroll
    for (int r = 0; r < 8; ++r) {
      float mx = fmaxf(s0[r], s1[r]);
#pragma unroll
      for (int off = 8; off >= 1; off >>= 1)
        mx = fmaxf(mx, __shfl_xor(mx, off, 16));
      float mn = fmaxf(mrow[r], mx);
      corr[r] = __expf(mrow[r] - mn);
      mrow[r] = mn;
      float p0 = __expf(s0[r] - mn);
      float p1 = __expf(s1[r] - mn);
      smem[wave][(r + 8 * hf) * 32 + ln]      = (h16)p0;
      smem[wave][(r + 8 * hf) * 32 + 16 + ln] = (h16)p1;
      ps[r] = p0 + p1;
    }
#pragma unroll
    for (int r = 0; r < 8; ++r) {
      float sm = ps[r];
#pragma unroll
      for (int off = 8; off >= 1; off >>= 1)
        sm += __shfl_xor(sm, off, 16);
      lrow[r] = lrow[r] * corr[r] + sm;
    }
#pragma unroll
    for (int d = 0; d < 4; ++d)
#pragma unroll
      for (int r = 0; r < 8; ++r) o[d][r] *= corr[r];

    // LDS writes are in-order with LDS reads per wave; fence for safety.
    asm volatile("s_wait_dscnt 0" ::: "memory");
    v16h pa = load_a(&smem[wave][0], 32, 0, 0, lane);   // ds_load x8

    // O += P (16x32) x V (32x64); Vt gives B^T layout directly.
    o[0] = wmma16(pa, load_b(Vb, TT, 0,  kb, lane), o[0]);
    o[1] = wmma16(pa, load_b(Vb, TT, 16, kb, lane), o[1]);
    o[2] = wmma16(pa, load_b(Vb, TT, 32, kb, lane), o[2]);
    o[3] = wmma16(pa, load_b(Vb, TT, 48, kb, lane), o[3]);
  }

  // Normalize and store Y in [B,T,H,D] so it is row-major [BT, C] for proj.
  const int b = bh >> 4, h = bh & 15;
#pragma unroll
  for (int r = 0; r < 8; ++r) {
    float inv = 1.0f / lrow[r];
    int t = qb + r + 8 * hf;
    size_t base = ((size_t)(b * TT + t) * HH + h) * DD + ln;
    Y[base + 0]  = (h16)(o[0][r] * inv);
    Y[base + 16] = (h16)(o[1][r] * inv);
    Y[base + 32] = (h16)(o[2][r] * inv);
    Y[base + 48] = (h16)(o[3][r] * inv);
  }
}

// ---------------------------------------------------------------------------
// Stage 4: output projection. [8192,1024] x [1024,1024] -> fp32 d_out.
// Same 32M x 64N wave tile as stage 2.
// ---------------------------------------------------------------------------
__global__ __launch_bounds__(256) void k_gemm_proj(
    const h16* __restrict__ Yh, const h16* __restrict__ Wt,
    float* __restrict__ Out) {
  const int lane = threadIdx.x & 31, wave = threadIdx.x >> 5;
  const int hf = lane >> 4, ln = lane & 15;
  const int m0 = blockIdx.y * 64  + (wave >> 2) * 32;
  const int n0 = blockIdx.x * 256 + (wave & 3) * 64;

  v8f acc[2][4] = {{v8f{}, v8f{}, v8f{}, v8f{}}, {v8f{}, v8f{}, v8f{}, v8f{}}};
  for (int k0 = 0; k0 < CC; k0 += 32) {
    if (k0 + 256 < CC) {
      __builtin_prefetch(Yh + (size_t)(m0 + ln) * CC + k0 + 256, 0, 3);
      __builtin_prefetch(Yh + (size_t)(m0 + 16 + ln) * CC + k0 + 256, 0, 3);
      __builtin_prefetch(Wt + (size_t)(n0 + hf * 32 + ln) * CC + k0 + 256, 0, 3);
    }
    v16h a0 = load_a(Yh, CC, m0,      k0, lane);
    v16h a1 = load_a(Yh, CC, m0 + 16, k0, lane);
#pragma unroll
    for (int j = 0; j < 4; ++j) {
      v16h b = load_b(Wt, CC, n0 + j * 16, k0, lane);
      acc[0][j] = wmma16(a0, b, acc[0][j]);
      acc[1][j] = wmma16(a1, b, acc[1][j]);
    }
  }
#pragma unroll
  for (int mi = 0; mi < 2; ++mi) {
#pragma unroll
    for (int j = 0; j < 4; ++j) {
      int col = n0 + j * 16 + ln;
#pragma unroll
      for (int r = 0; r < 8; ++r) {
        int row = m0 + mi * 16 + r + 8 * hf;
        Out[(size_t)row * CC + col] = acc[mi][j][r];
      }
    }
  }
}

// ---------------------------------------------------------------------------
// Host entry
// ---------------------------------------------------------------------------
extern "C" void kernel_launch(void* const* d_in, const int* in_sizes, int n_in,
                              void* d_out, int out_size, void* d_ws, size_t ws_size,
                              hipStream_t stream) {
  (void)in_sizes; (void)n_in; (void)out_size; (void)ws_size;
  const float* x      = (const float*)d_in[0];   // [4,2048,1024]
  const float* w_attn = (const float*)d_in[1];   // [1024,3072]
  const float* w_proj = (const float*)d_in[2];   // [1024,1024]
  float* out = (float*)d_out;                    // [4,2048,1024]

  // Workspace carve-up (f16 elements). Total ~88 MB.
  h16* p   = (h16*)d_ws;
  h16* xh  = p; p += (size_t)BT * CC;            // x in f16
  h16* waT = p; p += (size_t)CC * C3;            // w_attn^T  [3072,1024]
  h16* wpT = p; p += (size_t)CC * CC;            // w_proj^T  [1024,1024]
  h16* Qb  = p; p += (size_t)BT * CC;            // [BH,T,64]
  h16* Kb  = p; p += (size_t)BT * CC;            // [BH,T,64]
  h16* Vt  = p; p += (size_t)BT * CC;            // [BH,64,T]
  h16* Yb  = p;                                  // [B,T,H,D] == [BT,C]

  // Stage 1: casts / transposes
  k_cast_f32_f16<<<(BT * CC) / 256, 256, 0, stream>>>(x, xh, BT * CC);
  k_transpose_cast<<<(CC * C3 + 255) / 256, 256, 0, stream>>>(w_attn, waT, CC, C3);
  k_transpose_cast<<<(CC * CC + 255) / 256, 256, 0, stream>>>(w_proj, wpT, CC, CC);

  // Stage 2: QKV GEMM  (grid: N/256 x M/64)
  k_gemm_qkv<<<dim3(C3 / 256, BT / 64), 256, 0, stream>>>(xh, waT, Qb, Kb, Vt);

  // Stage 3: flash attention (8192 q-tiles, 8 waves/block)
  k_attn<<<(BB * HH * (TT / 16)) / 8, 256, 0, stream>>>(Qb, Kb, Vt, Yb);

  // Stage 4: projection GEMM -> fp32 output
  k_gemm_proj<<<dim3(CC / 256, BT / 64), 256, 0, stream>>>(Yb, wpT, out);
}
